// PointTransformerLayer_9165460209725
// MI455X (gfx1250) — matrix operand order repaired and checked by reading, not compiled, not run on hardware
//
#include <hip/hip_runtime.h>
#include <hip/hip_bf16.h>

typedef __attribute__((ext_vector_type(16))) _Float16 v16h;
typedef __attribute__((ext_vector_type(8)))  float    v8f;

constexpr int NPTS = 50000;
constexpr int C    = 128;
constexpr int KNEI = 16;
constexpr int CS   = 16;
constexpr float BN_EPS = 1e-5f;

__device__ __forceinline__ float4 ld4(const float* p) { return *(const float4*)p; }

// ---------------------------------------------------------------------------
// Kernel 1: q/k/v projections.  out[N,128] = feats[N,128] @ W[128,128] + b
// blockIdx.y selects which of the three weight sets.  256 threads = 8 waves,
// each wave owns a 16-row x 128-col tile; K accumulated in 4 chunks of 32
// via v_wmma_f32_16x16x32_f16.
// ---------------------------------------------------------------------------
__global__ __launch_bounds__(256) void qkv_gemm(
    const float* __restrict__ feats,
    const float* __restrict__ Wq, const float* __restrict__ bq,
    const float* __restrict__ Wk, const float* __restrict__ bk,
    const float* __restrict__ Wv, const float* __restrict__ bv,
    float* __restrict__ qo, float* __restrict__ ko, float* __restrict__ vo)
{
    __shared__ _Float16 s_wt[C][C];   // s_wt[col][k]  (transposed, f16)
    __shared__ float    s_b[C];

    const float* W; const float* bias; float* out;
    if (blockIdx.y == 0)      { W = Wq; bias = bq; out = qo; }
    else if (blockIdx.y == 1) { W = Wk; bias = bk; out = ko; }
    else                      { W = Wv; bias = bv; out = vo; }

    int tid = threadIdx.x;
    for (int i = tid; i < C * C; i += 256) {
        int k = i >> 7, c = i & (C - 1);
        s_wt[c][k] = (_Float16)W[i];            // W stored [in(k), out(c)]
    }
    for (int i = tid; i < C; i += 256) s_b[i] = bias[i];
    __syncthreads();

    int lane = tid & 31, wv = tid >> 5;
    int hi = lane >> 4, lj = lane & 15;
    int tilebase = blockIdx.x * 128 + wv * 16;
    int arow = tilebase + lj;                   // A row loaded by this lane
    int arowc = arow < NPTS ? arow : NPTS - 1;  // clamp for tail tile

    v8f acc[8];
    #pragma unroll
    for (int jt = 0; jt < 8; ++jt)
        acc[jt] = (v8f){0.f, 0.f, 0.f, 0.f, 0.f, 0.f, 0.f, 0.f};

    #pragma unroll
    for (int kc = 0; kc < C; kc += 32) {
        // A layout: lanes0-15 K = kc+{0..15}; lanes16-31 K = kc+{16..31}
        const float* fr = feats + (size_t)arowc * C + kc + 16 * hi;
        float4 x0 = ld4(fr), x1 = ld4(fr + 4), x2 = ld4(fr + 8), x3 = ld4(fr + 12);
        float av[16] = {x0.x, x0.y, x0.z, x0.w, x1.x, x1.y, x1.z, x1.w,
                        x2.x, x2.y, x2.z, x2.w, x3.x, x3.y, x3.z, x3.w};
        v16h A;
        #pragma unroll
        for (int e = 0; e < 16; ++e) A[e] = (_Float16)av[e];

        #pragma unroll
        for (int jt = 0; jt < 8; ++jt) {
            int col = jt * 16 + lj;
            // B layout: K = kc + e + 16*hi, contiguous in s_wt[col][]
            const _Float16* wp = &s_wt[col][kc + 16 * hi];
            v16h B;
            #pragma unroll
            for (int e = 0; e < 16; ++e) B[e] = wp[e];
            acc[jt] = __builtin_amdgcn_wmma_f32_16x16x32_f16(
                false, A, false, B, (short)0, acc[jt], false, false);
        }
    }

    // C/D layout: col = lane%16, rows = r + 8*hi
    #pragma unroll
    for (int jt = 0; jt < 8; ++jt) {
        int col = jt * 16 + lj;
        #pragma unroll
        for (int r = 0; r < 8; ++r) {
            int orow = tilebase + r + 8 * hi;
            if (orow < NPTS)
                out[(size_t)orow * C + col] = acc[jt][r] + s_b[col];
        }
    }
}

// ---------------------------------------------------------------------------
// Kernel 2: fused attention.  One wave (32 lanes) per point.
//   score pipeline is a 16x16x128 WMMA, second linear a K-padded 16x16 WMMA,
//   softmax over neighbors via one shfl_xor(16), grouped weighted sum VALU.
// ---------------------------------------------------------------------------
__global__ __launch_bounds__(128) void attn_kernel(
    const float* __restrict__ xyz, const float* __restrict__ feats,
    const int* __restrict__ nei,
    const float* __restrict__ q, const float* __restrict__ kf,
    const float* __restrict__ vf,
    const float* __restrict__ p1W, const float* __restrict__ p1b,
    const float* __restrict__ p1g, const float* __restrict__ p1be,
    const float* __restrict__ p2W, const float* __restrict__ p2b,
    const float* __restrict__ bnwg, const float* __restrict__ bnwb,
    const float* __restrict__ w1W, const float* __restrict__ w1b,
    const float* __restrict__ w1g, const float* __restrict__ w1be,
    const float* __restrict__ w2W, const float* __restrict__ w2b,
    float* __restrict__ out)
{
    __shared__ float    s_p1w[9], s_p1b[3], s_p1s[3], s_p1bb[3];
    __shared__ float    s_p2[3][C];
    __shared__ float    s_p2b[C];
    __shared__ float    s_bnws[C], s_bnwb[C];
    __shared__ _Float16 s_w1t[CS][C];          // [out col][in k]
    __shared__ float    s_w1b[CS], s_w1s[CS], s_w1bb[CS], s_w2b[CS];
    __shared__ _Float16 s_w2t[CS][CS];         // [out col][in cs]
    __shared__ float    s_buf[4][16][17];      // per-wave: h then softmax w
    __shared__ float    s_r3[4][16][3];        // per-wave pos-MLP hidden
    __shared__ int      s_idx[4][16];          // per-wave neighbor ids

    int tid = threadIdx.x;
    float inv = rsqrtf(1.f + BN_EPS);
    if (tid < 9) s_p1w[tid] = p1W[tid];
    if (tid < 3) { s_p1b[tid] = p1b[tid]; s_p1s[tid] = p1g[tid] * inv; s_p1bb[tid] = p1be[tid]; }
    for (int i = tid; i < 3 * C; i += 128) s_p2[i / C][i % C] = p2W[i];
    for (int i = tid; i < C; i += 128) {
        s_p2b[i] = p2b[i]; s_bnws[i] = bnwg[i] * inv; s_bnwb[i] = bnwb[i];
    }
    for (int i = tid; i < C * CS; i += 128) {
        int k = i / CS, j = i % CS;
        s_w1t[j][k] = (_Float16)w1W[i];        // w1W stored [C, CS]
    }
    if (tid < CS) {
        s_w1b[tid] = w1b[tid]; s_w1s[tid] = w1g[tid] * inv;
        s_w1bb[tid] = w1be[tid]; s_w2b[tid] = w2b[tid];
    }
    for (int i = tid; i < CS * CS; i += 128) {
        int k = i / CS, j = i % CS;
        s_w2t[j][k] = (_Float16)w2W[i];        // w2W stored [CS, CS]
    }
    __syncthreads();

    int wv = tid >> 5, lane = tid & 31;
    int hi = lane >> 4, lj = lane & 15;
    int n = blockIdx.x * 4 + wv;
    if (n >= NPTS) return;                     // 12500*4 == 50000, never taken

    // --- neighbor + position-MLP hidden (lane handles neighbor lj) ---------
    int id = nei[n * KNEI + lj];
    float px = xyz[(size_t)n * 3], py = xyz[(size_t)n * 3 + 1], pz = xyz[(size_t)n * 3 + 2];
    float dx = xyz[(size_t)id * 3] - px, dy = xyz[(size_t)id * 3 + 1] - py,
          dz = xyz[(size_t)id * 3 + 2] - pz;
    float r3[3];
    #pragma unroll
    for (int j = 0; j < 3; ++j) {
        float y = dx * s_p1w[j] + dy * s_p1w[3 + j] + dz * s_p1w[6 + j] + s_p1b[j];
        y = y * s_p1s[j] + s_p1bb[j];
        r3[j] = fmaxf(y, 0.f);
    }
    if (hi == 0) {
        s_idx[wv][lj] = id;
        s_r3[wv][lj][0] = r3[0]; s_r3[wv][lj][1] = r3[1]; s_r3[wv][lj][2] = r3[2];
    }

    // --- WMMA #1: relu(bn(kf - q + p)) [16x128] @ w1 [128x16] --------------
    v8f acc = (v8f){0.f, 0.f, 0.f, 0.f, 0.f, 0.f, 0.f, 0.f};
    #pragma unroll
    for (int kc = 0; kc < C; kc += 32) {
        v16h A, B;
        #pragma unroll
        for (int e = 0; e < 16; ++e) {
            int c = kc + 16 * hi + e;
            float pv = r3[0] * s_p2[0][c] + r3[1] * s_p2[1][c] + r3[2] * s_p2[2][c] + s_p2b[c];
            float w0 = kf[(size_t)id * C + c] - q[(size_t)n * C + c] + pv;
            float wb = w0 * s_bnws[c] + s_bnwb[c];
            A[e] = (_Float16)fmaxf(wb, 0.f);
        }
        const _Float16* wp = &s_w1t[lj][kc + 16 * hi];
        #pragma unroll
        for (int e = 0; e < 16; ++e) B[e] = wp[e];
        acc = __builtin_amdgcn_wmma_f32_16x16x32_f16(
            false, A, false, B, (short)0, acc, false, false);
    }

    // bn + relu -> h, staged to LDS (C/D layout: col lj, rows r+8*hi)
    #pragma unroll
    for (int r = 0; r < 8; ++r) {
        float x = acc[r] + s_w1b[lj];
        x = x * s_w1s[lj] + s_w1bb[lj];
        s_buf[wv][r + 8 * hi][lj] = fmaxf(x, 0.f);
    }
    asm volatile("s_wait_dscnt 0x0" ::: "memory");

    // --- WMMA #2: h [16x16] @ w2 [16x16], K padded to 32 -------------------
    v16h A2, B2;
    #pragma unroll
    for (int e = 0; e < 8; ++e) A2[e] = (_Float16)s_buf[wv][lj][e + 8 * hi];
    #pragma unroll
    for (int e = 8; e < 16; ++e) A2[e] = (_Float16)0.f;
    #pragma unroll
    for (int e = 0; e < 16; ++e) B2[e] = (e >= 8) ? (_Float16)0.f : s_w2t[lj][e + 8 * hi];
    v8f acc2 = (v8f){0.f, 0.f, 0.f, 0.f, 0.f, 0.f, 0.f, 0.f};
    acc2 = __builtin_amdgcn_wmma_f32_16x16x32_f16(
        false, A2, false, B2, (short)0, acc2, false, false);

    // --- softmax over the 16 neighbors (rows) per output column lj ---------
    float sv[8];
    float mx = -3.402823466e+38f;
    #pragma unroll
    for (int r = 0; r < 8; ++r) { sv[r] = acc2[r] + s_w2b[lj]; mx = fmaxf(mx, sv[r]); }
    mx = fmaxf(mx, __shfl_xor(mx, 16, 32));
    float sum = 0.f;
    #pragma unroll
    for (int r = 0; r < 8; ++r) { sv[r] = __expf(sv[r] - mx); sum += sv[r]; }
    sum += __shfl_xor(sum, 16, 32);
    float rs = 1.f / sum;
    #pragma unroll
    for (int r = 0; r < 8; ++r) s_buf[wv][r + 8 * hi][lj] = sv[r] * rs;
    asm volatile("s_wait_dscnt 0x0" ::: "memory");

    // --- grouped weighted sum + residual + LeakyReLU -----------------------
    int c = lane * 4;                          // 4 channels per lane
    float a0 = 0.f, a1 = 0.f, a2 = 0.f, a3 = 0.f;
    #pragma unroll
    for (int kk = 0; kk < 16; ++kk) {
        int idn = s_idx[wv][kk];
        float4 vv = ld4(vf + (size_t)idn * C + c);
        float rr0 = s_r3[wv][kk][0], rr1 = s_r3[wv][kk][1], rr2 = s_r3[wv][kk][2];
        int cb = c & 15;                       // c%16 for channel group index
        float g0 = s_buf[wv][kk][cb + 0];
        float g1 = s_buf[wv][kk][cb + 1];
        float g2 = s_buf[wv][kk][cb + 2];
        float g3 = s_buf[wv][kk][cb + 3];
        float p0 = rr0 * s_p2[0][c + 0] + rr1 * s_p2[1][c + 0] + rr2 * s_p2[2][c + 0] + s_p2b[c + 0];
        float p1 = rr0 * s_p2[0][c + 1] + rr1 * s_p2[1][c + 1] + rr2 * s_p2[2][c + 1] + s_p2b[c + 1];
        float p2 = rr0 * s_p2[0][c + 2] + rr1 * s_p2[1][c + 2] + rr2 * s_p2[2][c + 2] + s_p2b[c + 2];
        float p3 = rr0 * s_p2[0][c + 3] + rr1 * s_p2[1][c + 3] + rr2 * s_p2[2][c + 3] + s_p2b[c + 3];
        a0 += (vv.x + p0) * g0;
        a1 += (vv.y + p1) * g1;
        a2 += (vv.z + p2) * g2;
        a3 += (vv.w + p3) * g3;
    }
    float4 f4 = ld4(feats + (size_t)n * C + c);
    float z0 = a0 + f4.x, z1 = a1 + f4.y, z2 = a2 + f4.z, z3 = a3 + f4.w;
    float4 o;
    o.x = z0 >= 0.f ? z0 : 0.1f * z0;
    o.y = z1 >= 0.f ? z1 : 0.1f * z1;
    o.z = z2 >= 0.f ? z2 : 0.1f * z2;
    o.w = z3 >= 0.f ? z3 : 0.1f * z3;
    *(float4*)(out + (size_t)n * C + c) = o;
}

// ---------------------------------------------------------------------------
extern "C" void kernel_launch(void* const* d_in, const int* in_sizes, int n_in,
                              void* d_out, int out_size, void* d_ws, size_t ws_size,
                              hipStream_t stream)
{
    const float* xyz   = (const float*)d_in[0];
    const float* feats = (const float*)d_in[1];
    const int*   nei   = (const int*)d_in[2];
    const float* Wq = (const float*)d_in[3];  const float* bq = (const float*)d_in[4];
    const float* Wk = (const float*)d_in[5];  const float* bk = (const float*)d_in[6];
    const float* Wv = (const float*)d_in[7];  const float* bv = (const float*)d_in[8];
    const float* p1W  = (const float*)d_in[9];  const float* p1b  = (const float*)d_in[10];
    const float* p1g  = (const float*)d_in[11]; const float* p1be = (const float*)d_in[12];
    const float* p2W  = (const float*)d_in[13]; const float* p2b  = (const float*)d_in[14];
    const float* bnwg = (const float*)d_in[15]; const float* bnwb = (const float*)d_in[16];
    const float* w1W  = (const float*)d_in[17]; const float* w1b  = (const float*)d_in[18];
    const float* w1g  = (const float*)d_in[19]; const float* w1be = (const float*)d_in[20];
    const float* w2W  = (const float*)d_in[21]; const float* w2b  = (const float*)d_in[22];

    float* qbuf = (float*)d_ws;
    float* kbuf = qbuf + (size_t)NPTS * C;
    float* vbuf = kbuf + (size_t)NPTS * C;

    dim3 g1((NPTS + 127) / 128, 3);
    qkv_gemm<<<g1, 256, 0, stream>>>(feats, Wq, bq, Wk, bk, Wv, bv, qbuf, kbuf, vbuf);

    attn_kernel<<<(NPTS + 3) / 4, 128, 0, stream>>>(
        xyz, feats, nei, qbuf, kbuf, vbuf,
        p1W, p1b, p1g, p1be, p2W, p2b, bnwg, bnwb,
        w1W, w1b, w1g, w1be, w2W, w2b, (float*)d_out);
}
